// ClusterPolicyNetwork_20572893348176
// MI455X (gfx1250) — matrix operand order repaired and checked by reading, not compiled, not run on hardware
//
#include <hip/hip_runtime.h>
#include <hip/hip_bf16.h>
#include <math.h>

typedef __attribute__((ext_vector_type(16))) __bf16 v16bf;
typedef __attribute__((ext_vector_type(8)))  float  v8f;

#define DM 128
#define NTOK 1024

static __device__ __forceinline__ v8f wmma_bf16(v16bf a, v16bf b, v8f c) {
  // D = A(16x32) * B(32x16) + C, fp32 accumulate
  return __builtin_amdgcn_wmma_f32_16x16x32_bf16(false, a, false, b, (short)0, c, false, false);
}

// ---------------------------------------------------------------------------
// 1) QKV projection: [1024,128] @ [384,128]^T -> Q,K,V stored bf16
// ---------------------------------------------------------------------------
__global__ __launch_bounds__(256) void qkv_kernel(
    const float* __restrict__ x, const float* __restrict__ w,
    const float* __restrict__ b,
    __bf16* __restrict__ Qb, __bf16* __restrict__ Kb, __bf16* __restrict__ Vb) {
  int idx = blockIdx.x * 256 + threadIdx.x;
  if (idx >= NTOK * 3 * DM) return;
  int row = idx / (3 * DM);
  int col = idx % (3 * DM);
  const float* xr = x + (size_t)row * DM;
  const float* wr = w + (size_t)col * DM;
  float acc = b[col];
#pragma unroll 4
  for (int k = 0; k < DM; ++k) acc += xr[k] * wr[k];
  __bf16 v = (__bf16)acc;
  if (col < DM)            Qb[(size_t)row * DM + col]          = v;
  else if (col < 2 * DM)   Kb[(size_t)row * DM + (col - DM)]   = v;
  else                     Vb[(size_t)row * DM + (col - 2*DM)] = v;
}

// ---------------------------------------------------------------------------
// 2) Attention: one block per 16-query tile. WMMA QK^T, softmax, WMMA P@V.
//    Also accumulates head-averaged attention weights -> attn_out.
// ---------------------------------------------------------------------------
__global__ __launch_bounds__(256) void attn_kernel(
    const __bf16* __restrict__ Qb, const __bf16* __restrict__ Kb,
    const __bf16* __restrict__ Vb,
    float* __restrict__ ctxw, float* __restrict__ attn_out) {
  extern __shared__ char smem[];
  float* S      = (float*)smem;            // 16*1024
  float* Pacc   = S + 16 * 1024;           // 16*1024
  float* ctxred = Pacc + 16 * 1024;        // 8*16*32
  float* red    = ctxred + 8 * 16 * 32;    // 16*16

  const int tid   = threadIdx.x;
  const int lane  = tid & 31;
  const int wave  = tid >> 5;
  const int qbase = blockIdx.x * 16;
  const float scale = 0.17677669529663687f; // 1/sqrt(32)

  for (int i = tid; i < 16 * 1024; i += 256) Pacc[i] = 0.f;
  __syncthreads();

  for (int h = 0; h < 4; ++h) {
    // ---- A operand = Q tile (16x32 bf16) ----
    v16bf qa;
    {
      int m = lane & 15;
      int base = (lane < 16) ? 0 : 16;
      const __bf16* p = Qb + (size_t)(qbase + m) * DM + h * 32 + base;
#pragma unroll
      for (int e = 0; e < 8; ++e) { qa[e] = p[e]; qa[8 + e] = p[8 + e]; }
    }
    // ---- scores: each wave handles 8 key-tiles ----
    for (int kt = wave * 8; kt < wave * 8 + 8; ++kt) {
      int colk = lane & 15;
      int koff = (lane < 16) ? 0 : 16;
      const __bf16* p = Kb + (size_t)(kt * 16 + colk) * DM + h * 32 + koff;
      v16bf kb_;
#pragma unroll
      for (int e = 0; e < 16; ++e) kb_[e] = p[e];
      v8f acc = {0.f, 0.f, 0.f, 0.f, 0.f, 0.f, 0.f, 0.f};
      acc = wmma_bf16(qa, kb_, acc);
      int mo = (lane >= 16) ? 8 : 0;
#pragma unroll
      for (int j = 0; j < 8; ++j)
        S[(j + mo) * 1024 + kt * 16 + colk] = acc[j] * scale;
    }
    __syncthreads();
    // ---- softmax, 16 threads per row ----
    {
      int row = tid >> 4, sub = tid & 15;
      float lm = -1e30f;
      for (int i = sub; i < 1024; i += 16) lm = fmaxf(lm, S[row * 1024 + i]);
      red[row * 16 + sub] = lm;
      __syncthreads();
      float rm = red[row * 16];
#pragma unroll
      for (int i = 1; i < 16; ++i) rm = fmaxf(rm, red[row * 16 + i]);
      float ls = 0.f;
      for (int i = sub; i < 1024; i += 16) {
        float e = __expf(S[row * 1024 + i] - rm);
        S[row * 1024 + i] = e; ls += e;
      }
      __syncthreads();
      red[row * 16 + sub] = ls;
      __syncthreads();
      float rs = 0.f;
#pragma unroll
      for (int i = 0; i < 16; ++i) rs += red[row * 16 + i];
      float inv = 1.f / rs;
      for (int i = sub; i < 1024; i += 16) {
        float pv = S[row * 1024 + i] * inv;
        S[row * 1024 + i] = pv;
        Pacc[row * 1024 + i] += 0.25f * pv;
      }
    }
    __syncthreads();
    // ---- ctx = P @ V : each wave covers 128 keys (4 chunks of 32) ----
    v8f c0 = {0.f, 0.f, 0.f, 0.f, 0.f, 0.f, 0.f, 0.f};
    v8f c1 = {0.f, 0.f, 0.f, 0.f, 0.f, 0.f, 0.f, 0.f};
    for (int ch = 0; ch < 4; ++ch) {
      int keyb = wave * 128 + ch * 32;
      v16bf pa;
      {
        int m = lane & 15;
        int base = (lane < 16) ? 0 : 16;
#pragma unroll
        for (int e = 0; e < 8; ++e) {
          pa[e]     = (__bf16)S[m * 1024 + keyb + base + e];
          pa[8 + e] = (__bf16)S[m * 1024 + keyb + base + 8 + e];
        }
      }
      int colj = lane & 15;
      int koff = (lane < 16) ? 0 : 16;
      v16bf vb0, vb1;
#pragma unroll
      for (int e = 0; e < 16; ++e) {
        vb0[e] = Vb[(size_t)(keyb + koff + e) * DM + h * 32 + colj];
        vb1[e] = Vb[(size_t)(keyb + koff + e) * DM + h * 32 + 16 + colj];
      }
      c0 = wmma_bf16(pa, vb0, c0);
      c1 = wmma_bf16(pa, vb1, c1);
    }
    {
      int mo = (lane >= 16) ? 8 : 0;
      int n = lane & 15;
#pragma unroll
      for (int j = 0; j < 8; ++j) {
        ctxred[wave * 512 + (j + mo) * 32 + n]      = c0[j];
        ctxred[wave * 512 + (j + mo) * 32 + 16 + n] = c1[j];
      }
    }
    __syncthreads();
    for (int i = tid; i < 512; i += 256) {
      float s = 0.f;
#pragma unroll
      for (int w2 = 0; w2 < 8; ++w2) s += ctxred[w2 * 512 + i];
      int m = i >> 5, n = i & 31;
      ctxw[(size_t)(qbase + m) * DM + h * 32 + n] = s;
    }
    __syncthreads();
  }
  // head-averaged attention weights
  for (int i = tid; i < 16 * 1024; i += 256) {
    int m = i >> 10, c = i & 1023;
    attn_out[(size_t)(qbase + m) * 1024 + c] = Pacc[i];
  }
}

// ---------------------------------------------------------------------------
// 3) Output projection: attended = ctx @ out_w^T + out_b
// ---------------------------------------------------------------------------
__global__ __launch_bounds__(256) void outproj_kernel(
    const float* __restrict__ ctxw, const float* __restrict__ out_w,
    const float* __restrict__ out_b, float* __restrict__ attended) {
  int idx = blockIdx.x * 256 + threadIdx.x;
  if (idx >= NTOK * DM) return;
  int row = idx >> 7, col = idx & 127;
  const float* xr = ctxw + (size_t)row * DM;
  const float* wr = out_w + (size_t)col * DM;
  float acc = out_b[col];
#pragma unroll 4
  for (int k = 0; k < DM; ++k) acc += xr[k] * wr[k];
  attended[idx] = acc;
}

// ---------------------------------------------------------------------------
// 4) t1 = task @ W1[:, :128]^T ; n1 = attended @ W1[:, 128:]^T
// ---------------------------------------------------------------------------
__global__ __launch_bounds__(256) void t1n1_kernel(
    const float* __restrict__ task, const float* __restrict__ attended,
    const float* __restrict__ W1, float* __restrict__ t1,
    float* __restrict__ n1) {
  int idx = blockIdx.x * 256 + threadIdx.x;
  const int HALF = 1024 * 256;
  bool isN = idx >= HALF;
  int i = isN ? idx - HALF : idx;
  int row = i >> 8, c = i & 255;
  const float* xr = isN ? (attended + (size_t)row * DM) : (task + (size_t)row * DM);
  const float* wr = W1 + (size_t)c * 256 + (isN ? DM : 0);
  float acc = 0.f;
#pragma unroll 4
  for (int k = 0; k < DM; ++k) acc += xr[k] * wr[k];
  (isN ? n1 : t1)[i] = acc;
}

// ---------------------------------------------------------------------------
// 5) Coordination head: mean over rows, 2-layer MLP -> 32 logits
// ---------------------------------------------------------------------------
__global__ __launch_bounds__(256) void coord_kernel(
    const float* __restrict__ attended, const float* __restrict__ Wc1,
    const float* __restrict__ bc1, const float* __restrict__ Wc2,
    const float* __restrict__ bc2, float* __restrict__ coord_out) {
  __shared__ float gs[128];
  __shared__ float hid[256];
  int tid = threadIdx.x;
  if (tid < 128) {
    float s = 0.f;
    for (int i = 0; i < NTOK; ++i) s += attended[(size_t)i * DM + tid];
    gs[tid] = s * (1.0f / NTOK);
  }
  __syncthreads();
  {
    float h = bc1[tid];
    const float* wr = Wc1 + (size_t)tid * DM;
#pragma unroll 4
    for (int k = 0; k < DM; ++k) h += gs[k] * wr[k];
    hid[tid] = fmaxf(h, 0.f);
  }
  __syncthreads();
  if (tid < 32) {
    float o = bc2[tid];
    const float* wr = Wc2 + (size_t)tid * 256;
#pragma unroll 4
    for (int k = 0; k < 256; ++k) o += hid[k] * wr[k];
    coord_out[tid] = o;
  }
}

// ---------------------------------------------------------------------------
// 6) DOMINANT kernel: fused h1->h2->score. One block per (16 task x 16 node)
//    tile. M=256 pair-rows, K=256, N=128 via bf16 WMMA chains from LDS.
//    Register-blocked: waves form a 4x2 grid; each wave owns 4 M-tiles x
//    4 N-tiles with a live 4x4 v8f accumulator array. A loaded once per
//    K-step, B once per (ks,nt): ~1 ds_load_b128 per WMMA.
// ---------------------------------------------------------------------------
__global__ __launch_bounds__(256) void match_kernel(
    const float* __restrict__ t1, const float* __restrict__ n1,
    const float* __restrict__ b1, const float* __restrict__ W2,
    const float* __restrict__ b2, const float* __restrict__ W3,
    const float* __restrict__ b3, float* __restrict__ out) {
  extern __shared__ char smem[];
  const int H1S = 264;  // padded LDS row stride (elements) to avoid bank conflicts
  __bf16* h1b   = (__bf16*)smem;                               // 256 rows
  __bf16* w2b   = (__bf16*)(smem + (size_t)256 * H1S * 2);     // 128 rows
  float*  score = (float*)(smem + (size_t)256 * H1S * 2 + (size_t)128 * H1S * 2);

  const int tid  = threadIdx.x;
  const int lane = tid & 31;
  const int wave = tid >> 5;
  const int bt = blockIdx.x >> 6;
  const int bn = blockIdx.x & 63;

  // Stage W2 (bf16) in LDS
  for (int i = tid; i < 128 * 256; i += 256) {
    int r = i >> 8, c = i & 255;
    w2b[r * H1S + c] = (__bf16)W2[i];
  }
  // Build h1 tile = relu(t1[t] + n1[n] + b1) in LDS, bf16
  for (int i = tid; i < 256 * 256; i += 256) {
    int r = i >> 8, c = i & 255;
    int tl = r >> 4, nl = r & 15;
    float v = t1[(size_t)(bt * 16 + tl) * 256 + c] +
              n1[(size_t)(bn * 16 + nl) * 256 + c] + b1[c];
    h1b[r * H1S + c] = (__bf16)fmaxf(v, 0.f);
  }
  score[tid] = 0.f;
  __syncthreads();

  // wave grid: 4 (M) x 2 (N). Each wave: M-tiles [4*wm, 4*wm+3], N-tiles [4*wn, 4*wn+3]
  const int wm = wave & 3;
  const int wn = wave >> 2;
  const int mrow  = lane & 15;
  const int abase = (lane < 16) ? 0 : 16;
  const int bkoff = (lane < 16) ? 0 : 16;
  const int nlane = lane & 15;

  v8f acc[4][4];
#pragma unroll
  for (int i = 0; i < 4; ++i)
#pragma unroll
    for (int j = 0; j < 4; ++j)
      acc[i][j] = (v8f){0.f, 0.f, 0.f, 0.f, 0.f, 0.f, 0.f, 0.f};

#pragma unroll 2
  for (int ks = 0; ks < 8; ++ks) {
    int kb = ks * 32;
    v16bf A[4];
#pragma unroll
    for (int i = 0; i < 4; ++i) {
      const __bf16* pa = h1b + ((wm * 4 + i) * 16 + mrow) * H1S + kb + abase;
#pragma unroll
      for (int e = 0; e < 8; ++e) { A[i][e] = pa[e]; A[i][8 + e] = pa[8 + e]; }
    }
#pragma unroll
    for (int j = 0; j < 4; ++j) {
      int nt = wn * 4 + j;
      const __bf16* pb = w2b + (nt * 16 + nlane) * H1S + kb + bkoff;
      v16bf B;
#pragma unroll
      for (int e = 0; e < 16; ++e) B[e] = pb[e];
#pragma unroll
      for (int i = 0; i < 4; ++i)
        acc[i][j] = wmma_bf16(A[i], B, acc[i][j]);
    }
  }

  // fuse: relu(h2 + b2) * W3, reduce the 4 N-tiles in-register, then one
  // ds_add_f32 per (M-tile, row-slot) per lane.
  float b2v[4], w3v[4];
#pragma unroll
  for (int j = 0; j < 4; ++j) {
    int k = (wn * 4 + j) * 16 + nlane;
    b2v[j] = b2[k];
    w3v[j] = W3[k];
  }
  int mo = (lane >= 16) ? 8 : 0;
#pragma unroll
  for (int i = 0; i < 4; ++i) {
#pragma unroll
    for (int jj = 0; jj < 8; ++jj) {
      float s = 0.f;
#pragma unroll
      for (int j = 0; j < 4; ++j)
        s += fmaxf(acc[i][j][jj] + b2v[j], 0.f) * w3v[j];
      atomicAdd(&score[(wm * 4 + i) * 16 + jj + mo], s);
    }
  }
  __syncthreads();
  {
    float s = score[tid] + b3[0];
    float sig = 1.f / (1.f + __expf(-s));
    int tl = tid >> 4, nl = tid & 15;
    out[(size_t)(bt * 16 + tl) * 1024 + bn * 16 + nl] = sig;
  }
}

// ---------------------------------------------------------------------------
extern "C" void kernel_launch(void* const* d_in, const int* in_sizes, int n_in,
                              void* d_out, int out_size, void* d_ws, size_t ws_size,
                              hipStream_t stream) {
  (void)in_sizes; (void)n_in; (void)out_size; (void)ws_size;
  const float* node_states   = (const float*)d_in[0];
  const float* task_features = (const float*)d_in[1];
  const float* in_proj_w     = (const float*)d_in[2];
  const float* in_proj_b     = (const float*)d_in[3];
  const float* out_w         = (const float*)d_in[4];
  const float* out_b         = (const float*)d_in[5];
  const float* W1            = (const float*)d_in[6];
  const float* b1            = (const float*)d_in[7];
  const float* W2            = (const float*)d_in[8];
  const float* b2            = (const float*)d_in[9];
  const float* W3            = (const float*)d_in[10];
  const float* b3            = (const float*)d_in[11];
  const float* Wc1           = (const float*)d_in[12];
  const float* bc1           = (const float*)d_in[13];
  const float* Wc2           = (const float*)d_in[14];
  const float* bc2           = (const float*)d_in[15];

  char* wsb = (char*)d_ws;
  __bf16* Qb = (__bf16*)wsb;                     // 1024*128
  __bf16* Kb = Qb + 131072;
  __bf16* Vb = Kb + 131072;
  float* ctxw     = (float*)(wsb + 786432);       // 1024*128 f32
  float* attended = ctxw + 131072;                // 1024*128 f32
  float* t1       = attended + 131072;            // 1024*256 f32
  float* n1       = t1 + 262144;                  // 1024*256 f32

  float* outf      = (float*)d_out;
  float* match_out = outf;                        // [1024,1024]
  float* coord_out = outf + 1048576;              // [32]
  float* attn_out  = outf + 1048608;              // [1024,1024]

  qkv_kernel<<<1536, 256, 0, stream>>>(node_states, in_proj_w, in_proj_b, Qb, Kb, Vb);

  size_t attn_smem = (size_t)(16 * 1024 + 16 * 1024 + 8 * 16 * 32 + 16 * 16) * sizeof(float);
  attn_kernel<<<64, 256, attn_smem, stream>>>(Qb, Kb, Vb, ctxw, attn_out);

  outproj_kernel<<<512, 256, 0, stream>>>(ctxw, out_w, out_b, attended);
  t1n1_kernel<<<2048, 256, 0, stream>>>(task_features, attended, W1, t1, n1);
  coord_kernel<<<1, 256, 0, stream>>>(attended, Wc1, bc1, Wc2, bc2, coord_out);

  size_t match_smem = (size_t)256 * 264 * 2 + (size_t)128 * 264 * 2 + 256 * 4;
  match_kernel<<<4096, 256, match_smem, stream>>>(t1, n1, b1, W2, b2, W3, b3, match_out);
}
